// MultiHeadAttention_55628416418020
// MI455X (gfx1250) — matrix operand order repaired and checked by reading, not compiled, not run on hardware
//
#include <hip/hip_runtime.h>
#include <cstdint>

// ---------------------------------------------------------------------------
// CTRL-style MHA forward for MI455X (gfx1250, wave32, WMMA).
// B=2, S=2048, D_MODEL=1024, H=16, depth=64.
//
// Pipeline:
//   1) one-shot fp32->bf16 conversion of activations + weights (memory bound)
//   2) QKV projection GEMMs: pure-bf16 WMMA, W tiles async-staged to LDS
//      (double buffered, shared by all 8 waves of the block)
//   3) causal flash attention: K/V tiles async-staged to LDS (double
//      buffered, shared by 8 query-tile waves), P re-shaped via LDS
//   4) output GEMM (bf16 WMMA, fp32 out)
//
// Inner loops batch all B-fragment ds_loads first (single clause + single
// s_wait_dscnt), then run WMMAs back-to-back.
//
// Workspace (bf16 elems): Xq|Xk|Xv (3x4M) Wq|Wk|Wv|Wd (4x1M)
//                         Qh|Kh|Vt|Ctx (4x4M)  => 32M elems = 64 MB (L2-resident)
// ---------------------------------------------------------------------------

#define DEV_INLINE __device__ __forceinline__

typedef __attribute__((ext_vector_type(16))) __bf16 bf16x16;
typedef __attribute__((ext_vector_type(8)))  __bf16 bf16x8;
typedef __attribute__((ext_vector_type(8)))  float  f32x8;

constexpr int S_LEN  = 2048;
constexpr int DMODEL = 1024;
constexpr int NHEAD  = 16;
constexpr int DHEAD  = 64;

DEV_INLINE f32x8 wmma_bf16(bf16x16 a, bf16x16 b, f32x8 c) {
  return __builtin_amdgcn_wmma_f32_16x16x32_bf16(false, a, false, b,
                                                 (short)0, c, false, false);
}

// -------- CDNA5 async copy: global -> LDS, 16 bytes, tracked by ASYNCcnt ----
DEV_INLINE void async_load_b128(uint32_t lds_addr, const void* gptr) {
  asm volatile("global_load_async_to_lds_b128 %0, %1, off"
               :: "v"(lds_addr), "v"((uint64_t)(uintptr_t)gptr)
               : "memory");
}
DEV_INLINE void wait_async0() {
  asm volatile("s_wait_asynccnt 0" ::: "memory");
}
DEV_INLINE uint32_t lds_off(const void* p) {  // low 32 bits of generic = LDS offset
  return (uint32_t)(uintptr_t)p;
}

// ---- A fragment (16x32 bf16): lane<16 -> M=lane, K={0..7,16..23};
//      lane>=16 -> M=lane-16, K={8..15,24..31} (relative to k0)
DEV_INLINE bf16x16 load_A_bf16(const __bf16* X, int ld, int row0, int k0) {
  const int lane = threadIdx.x & 31;
  const int mrow = lane & 15;
  const int koff = (lane & 16) ? 8 : 0;
  const __bf16* p = X + (size_t)(row0 + mrow) * ld + k0 + koff;
  bf16x16 a;
#pragma unroll
  for (int i = 0; i < 8; ++i) { a[i] = p[i]; a[i + 8] = p[i + 16]; }
  return a;
}

// ---- B fragment (32x16 bf16): lane = N (mod 16); lane<16 holds K={0..7,16..23},
//      lane>=16 holds K={8..15,24..31} (relative to k0). Source: Bsrc[n][k] row-major.
DEV_INLINE bf16x16 load_B_lds(const __bf16* tile, int ld, int n0, int k0) {
  const int lane = threadIdx.x & 31;
  const int n = lane & 15;
  const int koff = (lane & 16) ? 8 : 0;
  const __bf16* p = tile + (n0 + n) * ld + k0 + koff;
  bf16x16 b;
#pragma unroll
  for (int i = 0; i < 8; ++i) { b[i] = p[i]; b[i + 8] = p[i + 16]; }
  return b;
}

// P-tile A fragment from per-wave LDS slice (16 rows x 32 cols bf16)
DEV_INLINE bf16x16 load_A_lds(const __bf16* lp) {
  const int lane = threadIdx.x & 31;
  const int mrow = lane & 15;
  const int koff = (lane & 16) ? 8 : 0;
  const __bf16* p = lp + mrow * 32 + koff;
  bf16x16 a;
#pragma unroll
  for (int i = 0; i < 8; ++i) { a[i] = p[i]; a[i + 8] = p[i + 16]; }
  return a;
}

// ===========================================================================
// Kernel 0: fp32 -> bf16 bulk conversion (8 elems / thread)
// ===========================================================================
__global__ void __launch_bounds__(256)
cvt_kernel(const float* __restrict__ in, __bf16* __restrict__ outp, int n8) {
  const int i = blockIdx.x * 256 + threadIdx.x;
  if (i >= n8) return;
  const float* p = in + (size_t)i * 8;
  bf16x8 r;
#pragma unroll
  for (int j = 0; j < 8; ++j) r[j] = (__bf16)p[j];
  *(bf16x8*)(outp + (size_t)i * 8) = r;
}

// ===========================================================================
// Kernel 1: projection GEMM  Y = X @ W^T + bias  (bf16 in -> bf16 out)
// layout 0: head-split [B,H,S,64]; layout 1: transposed [B,H,64,S]
// Wave tile 32x64; all 8 waves of a block share col0 -> W tile async-staged
// to LDS (64 rows x 32 k, 4 KB), double buffered.
// Grid: 2048 waves = 256 blocks x 8 waves.
// ===========================================================================
__global__ void __launch_bounds__(256)
proj_kernel(const __bf16* __restrict__ X, const __bf16* __restrict__ Wb,
            const float* __restrict__ bias, __bf16* __restrict__ Y, int layout) {
  __shared__ __bf16 wtile[2][64 * 32];

  const int lane = threadIdx.x & 31;
  const int wid = threadIdx.x >> 5;
  const int wglobal = blockIdx.x * 8 + wid;
  const int row0 = (wglobal & 127) * 32;   // 128 M-tiles of 32
  const int col0 = (wglobal >> 7) * 64;    // 16  N-tiles of 64 (uniform in block)

  // stage one 64x32 W tile: thread t copies 16B chunk (row t/4, k-chunk t%4)
  const int st_row = threadIdx.x >> 2;
  const int st_c   = (threadIdx.x & 3) * 8;

  async_load_b128(lds_off(&wtile[0][st_row * 32 + st_c]),
                  Wb + (size_t)(col0 + st_row) * DMODEL + 0 + st_c);

  f32x8 acc[2][4] = {};
  int buf = 0;
  for (int k0 = 0; k0 < DMODEL; k0 += 32, buf ^= 1) {
    wait_async0();
    __syncthreads();
    if (k0 + 32 < DMODEL)
      async_load_b128(lds_off(&wtile[buf ^ 1][st_row * 32 + st_c]),
                      Wb + (size_t)(col0 + st_row) * DMODEL + (k0 + 32) + st_c);

    // issue global A loads first, then batch all 4 B ds_loads, then 8 WMMAs
    bf16x16 a0 = load_A_bf16(X, DMODEL, row0, k0);
    bf16x16 a1 = load_A_bf16(X, DMODEL, row0 + 16, k0);
    bf16x16 bfr[4];
#pragma unroll
    for (int ni = 0; ni < 4; ++ni)
      bfr[ni] = load_B_lds(&wtile[buf][0], 32, ni * 16, 0);
#pragma unroll
    for (int ni = 0; ni < 4; ++ni) {
      acc[0][ni] = wmma_bf16(a0, bfr[ni], acc[0][ni]);
      acc[1][ni] = wmma_bf16(a1, bfr[ni], acc[1][ni]);
    }
  }

  const int off = (lane & 16) ? 8 : 0;
  const int cn = lane & 15;
#pragma unroll
  for (int ni = 0; ni < 4; ++ni) {
    const int c = col0 + ni * 16 + cn;
    const float bc = bias[c];
    const int h = c >> 6, d = c & 63;
#pragma unroll
    for (int mi = 0; mi < 2; ++mi) {
#pragma unroll
      for (int v = 0; v < 8; ++v) {
        const int r = row0 + mi * 16 + v + off;
        const int bb = r >> 11, s = r & (S_LEN - 1);
        const float val = acc[mi][ni][v] + bc;
        size_t idx;
        if (layout == 0)
          idx = (((size_t)(bb * NHEAD + h) * S_LEN + s) << 6) + d;
        else
          idx = ((size_t)(bb * NHEAD + h) * DHEAD + d) * S_LEN + s;
        Y[idx] = (__bf16)val;
      }
    }
  }
}

// ===========================================================================
// Kernel 2: causal flash attention. One wave = 16 query rows, depth 64.
// A block = 8 consecutive query tiles of one (b,h): the K/V 32-key blocks are
// shared, so they are async-staged to LDS (4 KB K + 4 KB V, double buffered).
// Block-uniform key loop (barriers legal); per-wave causal truncation guards
// the WMMA work wave-uniformly (EXEC stays all-ones).
// Grid: B*H*(S/16) = 4096 waves = 512 blocks x 8 waves. 24 KB LDS / block.
// ===========================================================================
__global__ void __launch_bounds__(256)
attn_kernel(const __bf16* __restrict__ Qh, const __bf16* __restrict__ Kh,
            const __bf16* __restrict__ Vt, __bf16* __restrict__ Ctx) {
  __shared__ __bf16 Kt[2][32 * 64];   // [key][depth]
  __shared__ __bf16 Vs[2][64 * 32];   // [depth][key]
  __shared__ __bf16 Pt[8][16 * 32];   // per-wave P staging

  const int lane = threadIdx.x & 31;
  const int wid = threadIdx.x >> 5;
  const int wglobal = blockIdx.x * 8 + wid;
  const int q0 = (wglobal & 127) * 16;
  const int bh = wglobal >> 7;                 // b*16 + h (uniform in block)
  const int qt_base = (blockIdx.x * 8) & 127;  // first q-tile of this block

  const __bf16* Qp = Qh + (size_t)bh * S_LEN * DHEAD;
  const __bf16* Kp = Kh + (size_t)bh * S_LEN * DHEAD;
  const __bf16* Vp = Vt + (size_t)bh * DHEAD * S_LEN;
  __bf16* lp = &Pt[wid][0];

  // staging assignment: K: thread t -> (key t/8, depth chunk t%8)
  //                     V: thread t -> (depth t/4, key chunk t%4)
  const int kr = threadIdx.x >> 3, kc = (threadIdx.x & 7) * 8;
  const int vr = threadIdx.x >> 2, vc = (threadIdx.x & 3) * 8;

  // Preload Q tile (16x64) as two A fragments
  bf16x16 qf0 = load_A_bf16(Qp, DHEAD, q0, 0);
  bf16x16 qf1 = load_A_bf16(Qp, DHEAD, q0, 32);

  f32x8 ctx[4] = {};
  float mstat[8], lstat[8];
#pragma unroll
  for (int v = 0; v < 8; ++v) { mstat[v] = -1e30f; lstat[v] = 0.0f; }

  const float scale = 0.125f;                   // 1/sqrt(64)
  const int off = (lane & 16) ? 8 : 0;
  const int cn = lane & 15;
  const int kend_w  = ((q0 + 16) + 31) & ~31;   // per-wave causal bound
  const int kend_blk = (qt_base + 8) * 16;      // block bound (mult of 128)

  async_load_b128(lds_off(&Kt[0][kr * 64 + kc]), Kp + (size_t)kr * DHEAD + kc);
  async_load_b128(lds_off(&Vs[0][vr * 32 + vc]), Vp + (size_t)vr * S_LEN + vc);

  int buf = 0;
  for (int key0 = 0; key0 < kend_blk; key0 += 32, buf ^= 1) {
    wait_async0();
    __syncthreads();
    if (key0 + 32 < kend_blk) {
      async_load_b128(lds_off(&Kt[buf ^ 1][kr * 64 + kc]),
                      Kp + (size_t)(key0 + 32 + kr) * DHEAD + kc);
      async_load_b128(lds_off(&Vs[buf ^ 1][vr * 32 + vc]),
                      Vp + (size_t)vr * S_LEN + (key0 + 32) + vc);
    }

    if (key0 < kend_w) {   // wave-uniform causal skip
      // ---- logits: batch 4 K-fragment ds_loads, then 4 WMMAs
      bf16x16 kb0 = load_B_lds(&Kt[buf][0], DHEAD, 0, 0);
      bf16x16 kb1 = load_B_lds(&Kt[buf][0], DHEAD, 0, 32);
      bf16x16 kb2 = load_B_lds(&Kt[buf][0], DHEAD, 16, 0);
      bf16x16 kb3 = load_B_lds(&Kt[buf][0], DHEAD, 16, 32);
      f32x8 s0 = {}, s1 = {};
      s0 = wmma_bf16(qf0, kb0, s0);
      s0 = wmma_bf16(qf1, kb1, s0);
      s1 = wmma_bf16(qf0, kb2, s1);
      s1 = wmma_bf16(qf1, kb3, s1);

      // ---- online softmax update, P -> LDS (C-layout -> A-layout re-shape)
      float alpha[8];
#pragma unroll
      for (int v = 0; v < 8; ++v) {
        const int qrow = q0 + v + off;
        float x0 = s0[v] * scale + ((key0 + cn      > qrow) ? -10000.0f : 0.0f);
        float x1 = s1[v] * scale + ((key0 + 16 + cn > qrow) ? -10000.0f : 0.0f);
        float rm = fmaxf(x0, x1);
#pragma unroll
        for (int i = 1; i < 16; i <<= 1) rm = fmaxf(rm, __shfl_xor(rm, i, 32));
        const float mnew = fmaxf(mstat[v], rm);
        const float a = __expf(mstat[v] - mnew);
        const float p0 = __expf(x0 - mnew);
        const float p1 = __expf(x1 - mnew);
        float rs = p0 + p1;
#pragma unroll
        for (int i = 1; i < 16; i <<= 1) rs += __shfl_xor(rs, i, 32);
        lstat[v] = lstat[v] * a + rs;
        mstat[v] = mnew;
        alpha[v] = a;
        const int prow = v + off;
        lp[prow * 32 + cn]      = (__bf16)p0;
        lp[prow * 32 + 16 + cn] = (__bf16)p1;
      }

#pragma unroll
      for (int df = 0; df < 4; ++df)
#pragma unroll
        for (int v = 0; v < 8; ++v) ctx[df][v] *= alpha[v];

      // drain per-wave LDS stores of P before re-reading (same-wave DS order)
      asm volatile("s_wait_dscnt 0" ::: "memory");

      // ---- ctx += P(16x32) @ V(32x64): batch 4 V-fragment loads + P, 4 WMMAs
      bf16x16 pf = load_A_lds(lp);
      bf16x16 vf[4];
#pragma unroll
      for (int df = 0; df < 4; ++df)
        vf[df] = load_B_lds(&Vs[buf][0], 32, df * 16, 0);
#pragma unroll
      for (int df = 0; df < 4; ++df)
        ctx[df] = wmma_bf16(pf, vf[df], ctx[df]);
    }
  }

  // ---- normalize + store ctx as bf16 [B,S,1024]
  float inv[8];
#pragma unroll
  for (int v = 0; v < 8; ++v) inv[v] = 1.0f / lstat[v];
  const int b = bh >> 4, h = bh & 15;
#pragma unroll
  for (int df = 0; df < 4; ++df) {
#pragma unroll
    for (int v = 0; v < 8; ++v) {
      const int s = q0 + v + off;
      const int c = h * DHEAD + df * 16 + cn;
      Ctx[((size_t)b * S_LEN + s) * DMODEL + c] = (__bf16)(ctx[df][v] * inv[v]);
    }
  }
}

// ===========================================================================
// Kernel 3: output GEMM  out = Ctx(bf16) @ dense_w^T + dense_b  -> fp32
// Same async-staged structure as proj_kernel.
// ===========================================================================
__global__ void __launch_bounds__(256)
out_kernel(const __bf16* __restrict__ Xb, const __bf16* __restrict__ Wb,
           const float* __restrict__ bias, float* __restrict__ Y) {
  __shared__ __bf16 wtile[2][64 * 32];

  const int lane = threadIdx.x & 31;
  const int wid = threadIdx.x >> 5;
  const int wglobal = blockIdx.x * 8 + wid;
  const int row0 = (wglobal & 127) * 32;
  const int col0 = (wglobal >> 7) * 64;

  const int st_row = threadIdx.x >> 2;
  const int st_c   = (threadIdx.x & 3) * 8;

  async_load_b128(lds_off(&wtile[0][st_row * 32 + st_c]),
                  Wb + (size_t)(col0 + st_row) * DMODEL + 0 + st_c);

  f32x8 acc[2][4] = {};
  int buf = 0;
  for (int k0 = 0; k0 < DMODEL; k0 += 32, buf ^= 1) {
    wait_async0();
    __syncthreads();
    if (k0 + 32 < DMODEL)
      async_load_b128(lds_off(&wtile[buf ^ 1][st_row * 32 + st_c]),
                      Wb + (size_t)(col0 + st_row) * DMODEL + (k0 + 32) + st_c);

    bf16x16 a0 = load_A_bf16(Xb, DMODEL, row0, k0);
    bf16x16 a1 = load_A_bf16(Xb, DMODEL, row0 + 16, k0);
    bf16x16 bfr[4];
#pragma unroll
    for (int ni = 0; ni < 4; ++ni)
      bfr[ni] = load_B_lds(&wtile[buf][0], 32, ni * 16, 0);
#pragma unroll
    for (int ni = 0; ni < 4; ++ni) {
      acc[0][ni] = wmma_bf16(a0, bfr[ni], acc[0][ni]);
      acc[1][ni] = wmma_bf16(a1, bfr[ni], acc[1][ni]);
    }
  }

  const int off = (lane & 16) ? 8 : 0;
  const int cn = lane & 15;
#pragma unroll
  for (int ni = 0; ni < 4; ++ni) {
    const int c = col0 + ni * 16 + cn;
    const float bc = bias[c];
#pragma unroll
    for (int mi = 0; mi < 2; ++mi) {
#pragma unroll
      for (int v = 0; v < 8; ++v) {
        const int r = row0 + mi * 16 + v + off;
        Y[(size_t)r * DMODEL + c] = acc[mi][ni][v] + bc;
      }
    }
  }
}

// ===========================================================================
extern "C" void kernel_launch(void* const* d_in, const int* in_sizes, int n_in,
                              void* d_out, int out_size, void* d_ws, size_t ws_size,
                              hipStream_t stream) {
  (void)in_sizes; (void)n_in; (void)out_size; (void)ws_size;
  const float* q       = (const float*)d_in[0];
  const float* k       = (const float*)d_in[1];
  const float* v       = (const float*)d_in[2];
  // d_in[3] = mask (causality applied analytically in-kernel)
  const float* Wq_w    = (const float*)d_in[4];
  const float* Wq_b    = (const float*)d_in[5];
  const float* Wk_w    = (const float*)d_in[6];
  const float* Wk_b    = (const float*)d_in[7];
  const float* Wv_w    = (const float*)d_in[8];
  const float* Wv_b    = (const float*)d_in[9];
  const float* dense_w = (const float*)d_in[10];
  const float* dense_b = (const float*)d_in[11];
  float* out = (float*)d_out;

  constexpr size_t E4M = (size_t)2 * S_LEN * DMODEL;  // 4,194,304
  constexpr size_t E1M = (size_t)DMODEL * DMODEL;     // 1,048,576

  __bf16* Xq  = (__bf16*)d_ws;
  __bf16* Xk  = Xq + E4M;
  __bf16* Xv  = Xk + E4M;
  __bf16* Wqb = Xv + E4M;
  __bf16* Wkb = Wqb + E1M;
  __bf16* Wvb = Wkb + E1M;
  __bf16* Wdb = Wvb + E1M;
  __bf16* Qh  = Wdb + E1M;
  __bf16* Kh  = Qh + E4M;
  __bf16* Vt  = Kh + E4M;
  __bf16* Ctx = Vt + E4M;

  // one-shot fp32 -> bf16 conversion (activations + weights)
  cvt_kernel<<<2048, 256, 0, stream>>>(q, Xq, (int)(E4M / 8));
  cvt_kernel<<<2048, 256, 0, stream>>>(k, Xk, (int)(E4M / 8));
  cvt_kernel<<<2048, 256, 0, stream>>>(v, Xv, (int)(E4M / 8));
  cvt_kernel<<<512, 256, 0, stream>>>(Wq_w, Wqb, (int)(E1M / 8));
  cvt_kernel<<<512, 256, 0, stream>>>(Wk_w, Wkb, (int)(E1M / 8));
  cvt_kernel<<<512, 256, 0, stream>>>(Wv_w, Wvb, (int)(E1M / 8));
  cvt_kernel<<<512, 256, 0, stream>>>(dense_w, Wdb, (int)(E1M / 8));

  // QKV projections (Q,K head-split; V transposed [B,H,64,S])
  proj_kernel<<<256, 256, 0, stream>>>(Xq, Wqb, Wq_b, Qh, 0);
  proj_kernel<<<256, 256, 0, stream>>>(Xk, Wkb, Wk_b, Kh, 0);
  proj_kernel<<<256, 256, 0, stream>>>(Xv, Wvb, Wv_b, Vt, 1);

  // causal flash attention
  attn_kernel<<<512, 256, 0, stream>>>(Qh, Kh, Vt, Ctx);

  // output projection
  out_kernel<<<256, 256, 0, stream>>>(Ctx, Wdb, dense_b, out);
}
